// VQVAE_20744692040088
// MI455X (gfx1250) — matrix operand order repaired and checked by reading, not compiled, not run on hardware
//
#include <hip/hip_runtime.h>
#include <math.h>

typedef __attribute__((ext_vector_type(16))) __bf16 v16bf;
typedef __attribute__((ext_vector_type(8)))  float  v8f;

union Frag { uint4 q[2]; v16bf v; };

static __device__ __forceinline__ void and4(uint4& u, unsigned m)
{
    u.x &= m; u.y &= m; u.z &= m; u.w &= m;
}

// ---------------------------------------------------------------------------
// Fast implicit-GEMM 3D conv / conv-transpose, WMMA bf16 16x16x32, f32 acc.
// Requires Cin % 32 == 0 (one K-chunk never crosses a conv tap).
// Activations bf16 NDHWC (fragment-native); weights bf16 [K/32][Cout][32].
// One wave computes a 32(M voxels) x 32(N channels) tile: two M sub-rows
// share B fragments -> 4 WMMA per 8 global_load_b128.
// epilogue: 0 bias+relu, 1 bias+swish, 2 bias+BN+PReLU+skip+relu,
//           3 bias+PReLU, 4 bias only.  outF32 selects f32 vs bf16 store.
// ---------------------------------------------------------------------------
template<int KS, int STRIDE, bool TRANS>
__global__ __launch_bounds__(256) void conv3d_fast_k(
    const __bf16* __restrict__ in, const __bf16* __restrict__ w2,
    const float* __restrict__ bias, void* __restrict__ outv,
    const __bf16* __restrict__ skip,
    const float* __restrict__ bng, const float* __restrict__ bnb,
    const float* __restrict__ bnm, const float* __restrict__ bnv,
    const float* __restrict__ alpha,
    int nBatch, int logInS, int logOutS, int logCin, int Cout,
    int pad, int mode, int tilesN, int outF32)
{
    const int inS  = 1 << logInS;
    const int outS = 1 << logOutS;
    const int lane = threadIdx.x & 31;
    const int wid  = (blockIdx.x * blockDim.x + threadIdx.x) >> 5;
    const int tn   = wid % tilesN;
    const int tm   = wid / tilesN;
    const int Mvox = nBatch << (3 * logOutS);
    if (tm * 32 >= Mvox) return;

    const int hi  = lane >> 4;
    const int l16 = lane & 15;

    // two M sub-rows: m0 = tm*32 + l16, m1 = m0 + 16 (same output x-row)
    int mrow = tm * 32 + l16;
    int ox0 = mrow & (outS - 1);
    int t  = mrow >> logOutS;
    int oy = t & (outS - 1);
    t >>= logOutS;
    int oz = t & (outS - 1);
    int ob = t >> logOutS;

    int bz, by, bx0, bx1;
    if (!TRANS) {
        bz = oz * STRIDE - pad; by = oy * STRIDE - pad;
        bx0 = ox0 * STRIDE - pad; bx1 = bx0 + 16 * STRIDE;
    } else {
        bz = oz + pad; by = oy + pad;
        bx0 = ox0 + pad; bx1 = bx0 + 16;
    }

    const int cinChunks = (1 << logCin) >> 5;   // Cin/32
    const int ch0 = tn * 32 + l16;
    const int ch1 = ch0 + 16;
    const bool fullN = (tn * 32 + 32) <= Cout;  // wave-uniform
    const int ch0c = (ch0 < Cout) ? ch0 : 0;
    const int ch1c = (ch1 < Cout) ? ch1 : 0;

    v8f acc00 = {}, acc01 = {}, acc10 = {}, acc11 = {};

    const int TAPS = KS * KS * KS;
    for (int tap = 0; tap < TAPS; ++tap) {
        const int kx = tap % KS;
        const int t2 = tap / KS;
        const int ky = t2 % KS;
        const int kz = t2 / KS;

        bool vzy, v0, v1;
        int rowbase = 0, ix0 = 0, ix1 = 0;
        if (!TRANS) {
            int iz = bz + kz, iy = by + ky;
            vzy = (unsigned)iz < (unsigned)inS && (unsigned)iy < (unsigned)inS;
            ix0 = bx0 + kx; ix1 = ix0 + 16 * STRIDE;
            v0 = vzy && (unsigned)ix0 < (unsigned)inS;
            v1 = vzy && (unsigned)ix1 < (unsigned)inS;
            if (vzy) rowbase = (((ob << logInS) | iz) << logInS | iy) << logInS;
        } else {
            int nz = bz - kz, ny = by - ky;
            int iz = nz >> 1, iy = ny >> 1;        // TRANS only used with STRIDE==2
            vzy = ((nz | ny) >= 0) && (((nz | ny) & 1) == 0) && iz < inS && iy < inS;
            int nx0 = bx0 - kx, nx1 = nx0 + 16;    // nx1 parity == nx0 parity? (+16 even) yes
            ix0 = nx0 >> 1; ix1 = nx1 >> 1;
            bool px = ((nx0 & 1) == 0);
            v0 = vzy && px && nx0 >= 0 && ix0 < inS;
            v1 = vzy && px && nx1 >= 0 && ix1 < inS;
            if (vzy) rowbase = (((ob << logInS) | iz) << logInS | iy) << logInS;
        }
        const int vox0 = v0 ? (rowbase | ix0) : 0;
        const int vox1 = v1 ? (rowbase | ix1) : 0;
        const unsigned msk0 = v0 ? 0xFFFFFFFFu : 0u;
        const unsigned msk1 = v1 ? 0xFFFFFFFFu : 0u;

        const uint4* ap0 = (const uint4*)(in + ((size_t)vox0 << logCin));
        const uint4* ap1 = (const uint4*)(in + ((size_t)vox1 << logCin));
        const __bf16* bp = w2 + (size_t)tap * cinChunks * Cout * 32;

        for (int c = 0; c < cinChunks; ++c) {
            // A fragments: halfs [c*32 + hi*8, +8) and [c*32 + 16 + hi*8, +8)
            uint4 a0a = ap0[4 * c + hi],     a0b = ap0[4 * c + 2 + hi];
            uint4 a1a = ap1[4 * c + hi],     a1b = ap1[4 * c + 2 + hi];
            and4(a0a, msk0); and4(a0b, msk0);
            and4(a1a, msk1); and4(a1b, msk1);
            Frag fa0; fa0.q[0] = a0a; fa0.q[1] = a0b;
            Frag fa1; fa1.q[0] = a1a; fa1.q[1] = a1b;

            // B fragments: halfs [hi*8, +8) and [16 + hi*8, +8) of repacked rows (kchunk, ch)
            const uint4* b0p = (const uint4*)(bp + ((size_t)c * Cout + ch0c) * 32);
            const uint4* b1p = (const uint4*)(bp + ((size_t)c * Cout + ch1c) * 32);
            uint4 b0a = b0p[hi], b0b = b0p[2 + hi];
            uint4 b1a = b1p[hi], b1b = b1p[2 + hi];
            if (!fullN) {
                unsigned m0 = (ch0 < Cout) ? 0xFFFFFFFFu : 0u;
                unsigned m1 = (ch1 < Cout) ? 0xFFFFFFFFu : 0u;
                and4(b0a, m0); and4(b0b, m0);
                and4(b1a, m1); and4(b1b, m1);
            }
            Frag fb0; fb0.q[0] = b0a; fb0.q[1] = b0b;
            Frag fb1; fb1.q[0] = b1a; fb1.q[1] = b1b;

            acc00 = __builtin_amdgcn_wmma_f32_16x16x32_bf16(false, fa0.v, false, fb0.v,
                                                            (short)0, acc00, false, false);
            acc01 = __builtin_amdgcn_wmma_f32_16x16x32_bf16(false, fa0.v, false, fb1.v,
                                                            (short)0, acc01, false, false);
            acc10 = __builtin_amdgcn_wmma_f32_16x16x32_bf16(false, fa1.v, false, fb0.v,
                                                            (short)0, acc10, false, false);
            acc11 = __builtin_amdgcn_wmma_f32_16x16x32_bf16(false, fa1.v, false, fb1.v,
                                                            (short)0, acc11, false, false);
        }
    }

    // epilogue: C/D layout VGPR r, lane -> M = r + 8*hi (within sub-row), N = l16
#pragma unroll
    for (int h = 0; h < 2; ++h) {
        int ch = tn * 32 + h * 16 + l16;
        if (ch >= Cout) continue;
        float bv = bias ? bias[ch] : 0.0f;
        float gg = 0.f, bb = 0.f, mm = 0.f, aa = 0.f, sc = 0.f;
        if (mode == 2) {
            gg = bng[ch]; bb = bnb[ch]; mm = bnm[ch]; aa = alpha[ch];
            sc = rsqrtf(bnv[ch] + 1e-3f);
        } else if (mode == 3) {
            aa = alpha[ch];
        }
#pragma unroll
        for (int g = 0; g < 2; ++g) {
            v8f acc = g ? (h ? acc11 : acc10) : (h ? acc01 : acc00);
#pragma unroll
            for (int r = 0; r < 8; ++r) {
                int ms = tm * 32 + g * 16 + r + (hi << 3);
                int addr = ms * Cout + ch;
                float v = acc[r] + bv;
                if (mode == 0)      v = fmaxf(v, 0.0f);
                else if (mode == 1) v = v / (1.0f + __expf(-v));
                else if (mode == 2) {
                    v = (v - mm) * gg * sc + bb;
                    v = v > 0.0f ? v : aa * v;
                    v += (float)skip[addr];
                    v = fmaxf(v, 0.0f);
                } else if (mode == 3) v = v > 0.0f ? v : aa * v;
                if (outF32) ((float*)outv)[addr] = v;
                else        ((__bf16*)outv)[addr] = (__bf16)v;
            }
        }
    }
}

// ---------------------------------------------------------------------------
// ec0: 4x4x4 stride-2 conv, Cin=1 (K = 64 taps), f32 input x, bf16 output.
// A is an element gather; B uses repacked bf16 weights.  bias+relu fused.
// ---------------------------------------------------------------------------
__global__ __launch_bounds__(256) void conv_ec0_k(
    const float* __restrict__ x, const __bf16* __restrict__ w2,
    const float* __restrict__ bias, __bf16* __restrict__ out, int tilesN)
{
    const int logInS = 7, logOutS = 6, Cout = 64, pad = 1;
    const int inS = 1 << logInS, outS = 1 << logOutS;
    const int lane = threadIdx.x & 31;
    const int wid  = (blockIdx.x * blockDim.x + threadIdx.x) >> 5;
    const int tn = wid % tilesN;
    const int tm = wid / tilesN;
    const int Mvox = 2 << (3 * logOutS);
    if (tm * 16 >= Mvox) return;
    const int hi = lane >> 4, l16 = lane & 15;

    int mrow = tm * 16 + l16;
    int ox = mrow & (outS - 1);
    int t = mrow >> logOutS;
    int oy = t & (outS - 1); t >>= logOutS;
    int oz = t & (outS - 1);
    int ob = t >> logOutS;
    int bz = oz * 2 - pad, by = oy * 2 - pad, bx = ox * 2 - pad;

    const int ch0 = tn * 32 + l16, ch1 = ch0 + 16;
    v8f acc0 = {}, acc1 = {};

    for (int k0 = 0; k0 < 64; k0 += 32) {
        v16bf afrag;
#pragma unroll
        for (int p = 0; p < 8; ++p) {
#pragma unroll
            for (int j = 0; j < 2; ++j) {
                const int koffA = ((p < 4) ? 0 : 16) + (hi << 3) + ((p & 3) << 1) + j;
                int tap = k0 + koffA;
                int kz = tap >> 4, ky = (tap >> 2) & 3, kx = tap & 3;
                int iz = bz + kz, iy = by + ky, ix = bx + kx;
                bool valid = (unsigned)iz < (unsigned)inS && (unsigned)iy < (unsigned)inS &&
                             (unsigned)ix < (unsigned)inS;
                int vox = valid ? ((((((ob << logInS) | iz) << logInS) | iy) << logInS) | ix) : 0;
                float av = x[vox];
                afrag[2 * p + j] = (__bf16)(valid ? av : 0.0f);
            }
        }
        const __bf16* bp = w2 + ((size_t)(k0 >> 5) * Cout) * 32;
        const uint4* b0p = (const uint4*)(bp + (size_t)ch0 * 32);
        const uint4* b1p = (const uint4*)(bp + (size_t)ch1 * 32);
        Frag fb0; fb0.q[0] = b0p[hi]; fb0.q[1] = b0p[2 + hi];
        Frag fb1; fb1.q[0] = b1p[hi]; fb1.q[1] = b1p[2 + hi];
        acc0 = __builtin_amdgcn_wmma_f32_16x16x32_bf16(false, afrag, false, fb0.v,
                                                       (short)0, acc0, false, false);
        acc1 = __builtin_amdgcn_wmma_f32_16x16x32_bf16(false, afrag, false, fb1.v,
                                                       (short)0, acc1, false, false);
    }

#pragma unroll
    for (int h = 0; h < 2; ++h) {
        int ch = tn * 32 + h * 16 + l16;
        v8f acc = h ? acc1 : acc0;
        float bv = bias[ch];
#pragma unroll
        for (int r = 0; r < 8; ++r) {
            int ms = tm * 16 + r + (hi << 3);
            float v = fmaxf(acc[r] + bv, 0.0f);
            out[ms * Cout + ch] = (__bf16)v;
        }
    }
}

// ---------------------------------------------------------------------------
// weight repack: f32 [K][Cout] -> bf16 [K/32][Cout][32]
// ---------------------------------------------------------------------------
__global__ void repack_w_k(const float* __restrict__ w, __bf16* __restrict__ w2,
                           int K, int Cout)
{
    int i = blockIdx.x * 256 + threadIdx.x;
    if (i >= K * Cout) return;
    int k = i / Cout, ch = i - k * Cout;
    w2[((size_t)(k >> 5) * Cout + ch) * 32 + (k & 31)] = (__bf16)w[(size_t)k * Cout + ch];
}

// ---------------------------------------------------------------------------
// VQ kernels
// ---------------------------------------------------------------------------
__global__ void vq_cnorm_k(const float* __restrict__ cb, float* __restrict__ cn, int K)
{
    int k = blockIdx.x * blockDim.x + threadIdx.x;
    if (k >= K) return;
    float s = 0.0f;
#pragma unroll 8
    for (int d = 0; d < 64; ++d) { float c = cb[k * 64 + d]; s += c * c; }
    cn[k] = s;
}

__global__ void vq_zero_k(float* __restrict__ c)
{
    if (threadIdx.x < 512) c[threadIdx.x] = 0.0f;
}

__global__ __launch_bounds__(256) void vq_assign_k(
    const __bf16* __restrict__ z, const float* __restrict__ cb,
    const float* __restrict__ cn, __bf16* __restrict__ zq,
    float* __restrict__ counts, int N)
{
    int i = blockIdx.x * 256 + threadIdx.x;
    if (i >= N) return;
    float zr[64];
#pragma unroll 16
    for (int d = 0; d < 64; ++d) zr[d] = (float)z[i * 64 + d];
    int best = 0;
    float bd = 3.4e38f;
    for (int k = 0; k < 512; ++k) {
        float dot = 0.0f;
#pragma unroll 16
        for (int d = 0; d < 64; ++d) dot = fmaf(zr[d], cb[k * 64 + d], dot);
        float dist = cn[k] - 2.0f * dot;       // +|z|^2 is argmin-invariant
        if (dist < bd) { bd = dist; best = k; }
    }
#pragma unroll 16
    for (int d = 0; d < 64; ++d) zq[i * 64 + d] = (__bf16)cb[best * 64 + d];
    atomicAdd(&counts[best], 1.0f);
}

__global__ void vq_perp_k(const float* __restrict__ counts, float* __restrict__ out, float invN)
{
    __shared__ float red[512];
    int t = threadIdx.x;
    float p = counts[t] * invN;
    red[t] = p * __logf(p + 1e-12f);
    __syncthreads();
    for (int s = 256; s > 0; s >>= 1) {
        if (t < s) red[t] += red[t + s];
        __syncthreads();
    }
    if (t == 0) out[0] = __expf(-red[0]);
}

// ---------------------------------------------------------------------------
// host-side helpers
// ---------------------------------------------------------------------------
template<int KS, int STRIDE, bool TRANS>
static void run_fast(hipStream_t st,
                     const __bf16* in, const __bf16* w2, const float* b, void* out,
                     const __bf16* skip, const float* g, const float* be,
                     const float* m, const float* v, const float* a,
                     int nB, int logInS, int logOutS, int logCin, int Cout,
                     int pad, int mode, int outF32)
{
    int tilesN = (Cout + 31) / 32;
    int Mvox = nB << (3 * logOutS);
    long waves = (long)(Mvox / 32) * tilesN;
    int blocks = (int)((waves + 7) / 8);
    conv3d_fast_k<KS, STRIDE, TRANS><<<blocks, 256, 0, st>>>(
        in, w2, b, out, skip, g, be, m, v, a,
        nB, logInS, logOutS, logCin, Cout, pad, mode, tilesN, outF32);
}

static void run_repack(hipStream_t st, const float* w, __bf16* w2, int K, int Cout)
{
    int n = K * Cout;
    repack_w_k<<<(n + 255) / 256, 256, 0, st>>>(w, w2, K, Cout);
}

extern "C" void kernel_launch(void* const* d_in, const int* in_sizes, int n_in,
                              void* d_out, int out_size, void* d_ws, size_t ws_size,
                              hipStream_t stream)
{
    (void)in_sizes; (void)n_in; (void)out_size; (void)ws_size;
#define FIN(i) ((const float*)d_in[(i)])
    const float* x      = FIN(0);
    const float* ec0_w  = FIN(1);  const float* ec0_b  = FIN(2);
    const float* er0_c1w= FIN(3);  const float* er0_c1b= FIN(4);
    const float* er0_c2w= FIN(5);  const float* er0_c2b= FIN(6);
    const float* er0_g  = FIN(7);  const float* er0_be = FIN(8);
    const float* er0_m  = FIN(9);  const float* er0_v  = FIN(10);
    const float* er0_a  = FIN(11);
    const float* ec1_w  = FIN(12); const float* ec1_b  = FIN(13);
    const float* er1_c1w= FIN(14); const float* er1_c1b= FIN(15);
    const float* er1_c2w= FIN(16); const float* er1_c2b= FIN(17);
    const float* er1_g  = FIN(18); const float* er1_be = FIN(19);
    const float* er1_m  = FIN(20); const float* er1_v  = FIN(21);
    const float* er1_a  = FIN(22);
    const float* eo_w   = FIN(23); const float* eo_b   = FIN(24); const float* eo_a = FIN(25);
    const float* codebook = FIN(26);
    const float* dc0_w  = FIN(27); const float* dc0_b  = FIN(28); const float* dc0_a = FIN(29);
    const float* dr0_c1w= FIN(30); const float* dr0_c1b= FIN(31);
    const float* dr0_c2w= FIN(32); const float* dr0_c2b= FIN(33);
    const float* dr0_g  = FIN(34); const float* dr0_be = FIN(35);
    const float* dr0_m  = FIN(36); const float* dr0_v  = FIN(37);
    const float* dr0_a  = FIN(38);
    const float* du0_w  = FIN(39); const float* du0_b  = FIN(40);
    const float* dr1_c1w= FIN(41); const float* dr1_c1b= FIN(42);
    const float* dr1_c2w= FIN(43); const float* dr1_c2b= FIN(44);
    const float* dr1_g  = FIN(45); const float* dr1_be = FIN(46);
    const float* dr1_m  = FIN(47); const float* dr1_v  = FIN(48);
    const float* dr1_a  = FIN(49);
    const float* du1_w  = FIN(50); const float* du1_b  = FIN(51);
#undef FIN

    // ---- bf16 activation arena ----
    char* ws = (char*)d_ws;
    const size_t B64     = (size_t)2 * 64 * 64 * 64 * 64 * 2;    // 64 MiB
    const size_t B32_128 = (size_t)2 * 32 * 32 * 32 * 128 * 2;   // 16 MiB
    const size_t B32_64  = (size_t)2 * 32 * 32 * 32 * 64 * 2;    //  8 MiB
    __bf16* bufA = (__bf16*)(ws);
    __bf16* bufB = (__bf16*)(ws + B64);
    __bf16* bufC = (__bf16*)(ws + 2 * B64);
    __bf16* s0   = (__bf16*)(ws + 3 * B64);
    __bf16* s1   = (__bf16*)(ws + 3 * B64 + B32_128);
    __bf16* s2   = (__bf16*)(ws + 3 * B64 + 2 * B32_128);
    __bf16* zlat = (__bf16*)(ws + 3 * B64 + 3 * B32_128);
    __bf16* zq   = (__bf16*)(ws + 3 * B64 + 3 * B32_128 + B32_64);
    float*  cnrm = (float*) (ws + 3 * B64 + 3 * B32_128 + 2 * B32_64);
    float*  cnts = (float*) (ws + 3 * B64 + 3 * B32_128 + 2 * B32_64 + 2048);
    size_t wcur  = 3 * B64 + 3 * B32_128 + 2 * B32_64 + 4096;
    auto walloc = [&](size_t elems) {
        __bf16* p = (__bf16*)(ws + wcur);
        wcur = (wcur + elems * 2 + 63) & ~(size_t)63;
        return p;
    };

    // ---- repack all weights to bf16 fragment layout ----
    __bf16* W_ec0  = walloc(64   * 64);  run_repack(stream, ec0_w,  W_ec0,  64,   64);
    __bf16* W_e0c1 = walloc(1728 * 64);  run_repack(stream, er0_c1w,W_e0c1, 1728, 64);
    __bf16* W_e0c2 = walloc(1728 * 64);  run_repack(stream, er0_c2w,W_e0c2, 1728, 64);
    __bf16* W_ec1  = walloc(4096 * 128); run_repack(stream, ec1_w,  W_ec1,  4096, 128);
    __bf16* W_e1c1 = walloc(3456 * 128); run_repack(stream, er1_c1w,W_e1c1, 3456, 128);
    __bf16* W_e1c2 = walloc(3456 * 128); run_repack(stream, er1_c2w,W_e1c2, 3456, 128);
    __bf16* W_eo   = walloc(3456 * 64);  run_repack(stream, eo_w,   W_eo,   3456, 64);
    __bf16* W_dc0  = walloc(1728 * 128); run_repack(stream, dc0_w,  W_dc0,  1728, 128);
    __bf16* W_d0c1 = walloc(3456 * 128); run_repack(stream, dr0_c1w,W_d0c1, 3456, 128);
    __bf16* W_d0c2 = walloc(3456 * 128); run_repack(stream, dr0_c2w,W_d0c2, 3456, 128);
    __bf16* W_du0  = walloc(8192 * 64);  run_repack(stream, du0_w,  W_du0,  8192, 64);
    __bf16* W_d1c1 = walloc(1728 * 64);  run_repack(stream, dr1_c1w,W_d1c1, 1728, 64);
    __bf16* W_d1c2 = walloc(1728 * 64);  run_repack(stream, dr1_c2w,W_d1c2, 1728, 64);
    __bf16* W_du1  = walloc(4096 * 1);   run_repack(stream, du1_w,  W_du1,  4096, 1);

    const float*  NF = nullptr;
    const __bf16* NB = nullptr;

    // =================== Encoder ===================
    // ec0: 4x4x4 s2, 1->64, relu          x[2,128^3,1] -> bufA[2,64^3,64]
    {
        long waves = (long)(524288 / 16) * 2;
        conv_ec0_k<<<(int)((waves + 7) / 8), 256, 0, stream>>>(x, W_ec0, ec0_b, bufA, 2);
    }
    // er0 conv1: swish
    run_fast<3, 1, false>(stream, bufA, W_e0c1, er0_c1b, bufB, NB, NF, NF, NF, NF, NF,
                          2, 6, 6, 6, 64, 1, 1, 0);
    // er0 conv2: BN+PReLU+skip(bufA)+relu
    run_fast<3, 1, false>(stream, bufB, W_e0c2, er0_c2b, bufC, bufA,
                          er0_g, er0_be, er0_m, er0_v, er0_a, 2, 6, 6, 6, 64, 1, 2, 0);
    // ec1: 4x4x4 s2, 64->128, relu        -> s0[2,32^3,128]
    run_fast<4, 2, false>(stream, bufC, W_ec1, ec1_b, s0, NB, NF, NF, NF, NF, NF,
                          2, 6, 5, 6, 128, 1, 0, 0);
    // er1 conv1: swish
    run_fast<3, 1, false>(stream, s0, W_e1c1, er1_c1b, s1, NB, NF, NF, NF, NF, NF,
                          2, 5, 5, 7, 128, 1, 1, 0);
    // er1 conv2: BN+PReLU+skip(s0)+relu
    run_fast<3, 1, false>(stream, s1, W_e1c2, er1_c2b, s2, s0,
                          er1_g, er1_be, er1_m, er1_v, er1_a, 2, 5, 5, 7, 128, 1, 2, 0);
    // eo: 3x3x3 128->64, PReLU -> latent
    run_fast<3, 1, false>(stream, s2, W_eo, eo_b, zlat, NB, NF, NF, NF, NF, eo_a,
                          2, 5, 5, 7, 64, 1, 3, 0);

    // =================== NSVQ (hard lookup) ===================
    vq_cnorm_k<<<2, 256, 0, stream>>>(codebook, cnrm, 512);
    vq_zero_k<<<1, 512, 0, stream>>>(cnts);
    vq_assign_k<<<65536 / 256, 256, 0, stream>>>(zlat, codebook, cnrm, zq, cnts, 65536);

    // =================== Decoder ===================
    // dc0: 3x3x3 64->128, PReLU
    run_fast<3, 1, false>(stream, zq, W_dc0, dc0_b, s0, NB, NF, NF, NF, NF, dc0_a,
                          2, 5, 5, 6, 128, 1, 3, 0);
    // dr0 conv1: swish
    run_fast<3, 1, false>(stream, s0, W_d0c1, dr0_c1b, s1, NB, NF, NF, NF, NF, NF,
                          2, 5, 5, 7, 128, 1, 1, 0);
    // dr0 conv2: BN+PReLU+skip(s0)+relu
    run_fast<3, 1, false>(stream, s1, W_d0c2, dr0_c2b, s2, s0,
                          dr0_g, dr0_be, dr0_m, dr0_v, dr0_a, 2, 5, 5, 7, 128, 1, 2, 0);
    // du0: convT 4x4x4 s2, 128->64, relu  -> bufA[2,64^3,64]
    run_fast<4, 2, true>(stream, s2, W_du0, du0_b, bufA, NB, NF, NF, NF, NF, NF,
                         2, 5, 6, 7, 64, 1, 0, 0);
    // dr1 conv1: swish
    run_fast<3, 1, false>(stream, bufA, W_d1c1, dr1_c1b, bufB, NB, NF, NF, NF, NF, NF,
                          2, 6, 6, 6, 64, 1, 1, 0);
    // dr1 conv2: BN+PReLU+skip(bufA)+relu
    run_fast<3, 1, false>(stream, bufB, W_d1c2, dr1_c2b, bufC, bufA,
                          dr1_g, dr1_be, dr1_m, dr1_v, dr1_a, 2, 6, 6, 6, 64, 1, 2, 0);
    // du1: convT 4x4x4 s2, 64->1, bias only -> f32 recon into d_out
    run_fast<4, 2, true>(stream, bufC, W_du1, du1_b, d_out, NB, NF, NF, NF, NF, NF,
                         2, 6, 7, 6, 1, 1, 4, 1);

    // perplexity -> d_out[2*128^3]
    vq_perp_k<<<1, 512, 0, stream>>>(cnts, ((float*)d_out) + 4194304, 1.0f / 65536.0f);
}
